// MN4_47124381172173
// MI455X (gfx1250) — matrix operand order repaired and checked by reading, not compiled, not run on hardware
//
#include <hip/hip_runtime.h>
#include <math.h>

// Problem constants (from reference)
#define B_   4
#define Q_   75
#define C_   640
#define M_   196            // h*w
#define MP_  208            // M padded to multiple of 16
#define NW_  5              // N_WAY
#define KS_  5              // K_SHOT
#define KCH_ 32             // WMMA K per step (f16)
#define TILES_ 13           // MP_/16
#define KSTEPS_ (C_/KCH_)   // 20
#define LROW_ 40            // LDS row pitch in halves (padded 32 -> 40, conflict-free)

typedef __attribute__((ext_vector_type(16))) _Float16 v16h;
typedef __attribute__((ext_vector_type(8)))  float    v8f;

union Frag {
  v16h h;
  int4 q[2];
};

// ---------------------------------------------------------------------------
// Kernel 1: k-shot mean + L2-normalize over channels + transpose -> f16
// out[b*NW+n][m][c], rows m in [196,208) zero-padded.
// ---------------------------------------------------------------------------
__global__ void mn4_prep_support(const float* __restrict__ sup, _Float16* __restrict__ out)
{
  const int bn  = blockIdx.x;                 // b*NW_ + n
  const int tid = threadIdx.x;
  const float* base = sup + (size_t)bn * KS_ * C_ * M_;   // shots contiguous
  __shared__ float invn[MP_];
  if (tid < MP_) {
    if (tid < M_) {
      float s = 0.f;
      for (int c = 0; c < C_; ++c) {
        float v = 0.f;
#pragma unroll
        for (int k = 0; k < KS_; ++k) v += base[((size_t)k * C_ + c) * M_ + tid];
        v *= 0.2f;
        s += v * v;
      }
      invn[tid] = 1.0f / fmaxf(sqrtf(s), 1e-8f);
    } else {
      invn[tid] = 0.f;
    }
  }
  __syncthreads();
  _Float16* o = out + (size_t)bn * MP_ * C_;
  for (int idx = tid; idx < MP_ * C_; idx += blockDim.x) {
    const int m = idx / C_, c = idx % C_;
    float v = 0.f;
    if (m < M_) {
#pragma unroll
      for (int k = 0; k < KS_; ++k) v += base[((size_t)k * C_ + c) * M_ + m];
      v = v * 0.2f * invn[m];
    }
    o[idx] = (_Float16)v;
  }
}

// ---------------------------------------------------------------------------
// Kernel 2: L2-normalize queries over channels + transpose -> f16
// ---------------------------------------------------------------------------
__global__ void mn4_prep_query(const float* __restrict__ qf, _Float16* __restrict__ out)
{
  const int bq  = blockIdx.x;                 // b*Q_ + q
  const int tid = threadIdx.x;
  const float* base = qf + (size_t)bq * C_ * M_;
  __shared__ float invn[MP_];
  if (tid < MP_) {
    if (tid < M_) {
      float s = 0.f;
      for (int c = 0; c < C_; ++c) { const float v = base[(size_t)c * M_ + tid]; s += v * v; }
      invn[tid] = 1.0f / fmaxf(sqrtf(s), 1e-8f);
    } else {
      invn[tid] = 0.f;
    }
  }
  __syncthreads();
  _Float16* o = out + (size_t)bq * MP_ * C_;
  for (int idx = tid; idx < MP_ * C_; idx += blockDim.x) {
    const int m = idx / C_, c = idx % C_;
    const float v = (m < M_) ? base[(size_t)c * M_ + m] * invn[m] : 0.f;
    o[idx] = (_Float16)v;
  }
}

// ---------------------------------------------------------------------------
// Kernel 3: per (b,q,n): S = Qn (196x640) * Pn^T (640x196) via WMMA f16->f32,
// then per row m: max_s S[m,s] and first-occurrence argmax.
// One workgroup = 512 threads = 16 wave32 (13 own a 16-row strip of S).
// K staged through LDS in chunks of 32 (padded pitch for conflict-free b128).
// ---------------------------------------------------------------------------
__global__ __launch_bounds__(512) void mn4_gemm(const _Float16* __restrict__ qn,
                                                const _Float16* __restrict__ sp,
                                                float* __restrict__ maxv,
                                                int*   __restrict__ args)
{
  const int id = blockIdx.x;            // bq*NW_ + n
  const int n  = id % NW_;
  const int bq = id / NW_;
  const int b  = bq / Q_;
  const _Float16* Abase = qn + (size_t)bq * MP_ * C_;
  const _Float16* Bbase = sp + (size_t)(b * NW_ + n) * MP_ * C_;
  float* mout = maxv + (size_t)id * M_;
  int*   aout = args + (size_t)id * M_;

  __shared__ _Float16 lA[MP_ * LROW_];  // 16640 B
  __shared__ _Float16 lB[MP_ * LROW_];  // 16640 B

  const int tid   = threadIdx.x;
  const int wave  = tid >> 5;
  const int lane  = tid & 31;
  const int hl    = lane & 15;
  const int upper = lane >> 4;          // 0: lanes 0-15, 1: lanes 16-31

  const int  ti     = wave;             // row-tile owned by this wave
  const bool active = (ti < TILES_);    // wave-uniform

  // Pre-compute this thread's staging role: one 64B row-chunk per thread.
  const bool stg   = (tid < 2 * MP_);
  const int  sarr  = (tid >= MP_) ? 1 : 0;
  const int  srow  = sarr ? (tid - MP_) : tid;
  const _Float16* sbase = (sarr ? Bbase : Abase) + (size_t)srow * C_;
  _Float16*       sdst  = (sarr ? lB : lA) + srow * LROW_;

  v8f acc[TILES_] = {};

  for (int ks = 0; ks < KSTEPS_; ++ks) {
    const int k0 = ks * KCH_;
    __syncthreads();                    // previous chunk fully consumed

    // Cooperative stage: 4 contiguous global_load_b128 then 4 ds_store_b128.
    if (stg) {
      const _Float16* src = sbase + k0;
      const int4 t0 = ((const int4*)src)[0];
      const int4 t1 = ((const int4*)src)[1];
      const int4 t2 = ((const int4*)src)[2];
      const int4 t3 = ((const int4*)src)[3];
      const int kpf = (k0 + KCH_ < C_) ? (k0 + KCH_) : k0;   // branchless clamp
      __builtin_prefetch(sbase + kpf, 0, 0);                  // next K-chunk
      ((int4*)sdst)[0] = t0;
      ((int4*)sdst)[1] = t1;
      ((int4*)sdst)[2] = t2;
      ((int4*)sdst)[3] = t3;
    }
    __syncthreads();

    if (active) {
      // A fragment (16x32 f16): lanes 0-15 hold K 0..15, lanes 16-31 K 16..31
      Frag af;
      const int arow = ti * 16 + hl;
      const int aoff = upper * 16;
      af.q[0] = *(const int4*)&lA[arow * LROW_ + aoff];
      af.q[1] = *(const int4*)&lA[arow * LROW_ + aoff + 8];
      const int boff = upper * 16;      // B: lanes 0-15 K 0..15, lanes 16-31 K 16..31

      // Double-buffered B fragments: tile tj+1's loads issued before wmma(tj).
      Frag bf[2];
      bf[0].q[0] = *(const int4*)&lB[hl * LROW_ + boff];
      bf[0].q[1] = *(const int4*)&lB[hl * LROW_ + boff + 8];
#pragma unroll
      for (int tj = 0; tj < TILES_; ++tj) {
        const int cur = tj & 1, nxt = cur ^ 1;
        if (tj + 1 < TILES_) {
          const int nrow = (tj + 1) * 16 + hl;
          bf[nxt].q[0] = *(const int4*)&lB[nrow * LROW_ + boff];
          bf[nxt].q[1] = *(const int4*)&lB[nrow * LROW_ + boff + 8];
        }
        acc[tj] = __builtin_amdgcn_wmma_f32_16x16x32_f16(
            false, af.h, false, bf[cur].h, (short)0, acc[tj], false, false);
      }

      // Pin the software pipeline: [4 DS-reads][2 DS-reads, 1 WMMA]x12 [1 WMMA].
      // 0x100 = DS read, 0x008 = MFMA/WMMA.
      __builtin_amdgcn_sched_group_barrier(0x100, 4, 0);
#pragma unroll
      for (int g = 0; g < TILES_ - 1; ++g) {
        __builtin_amdgcn_sched_group_barrier(0x100, 2, 0);
        __builtin_amdgcn_sched_group_barrier(0x008, 1, 0);
      }
      __builtin_amdgcn_sched_group_barrier(0x008, 1, 0);
    }
  }

  // Per-row max/argmax over s (< 196), first-occurrence tie-break.
  if (active) {
#pragma unroll
    for (int r = 0; r < 8; ++r) {
      float best = -INFINITY; int bs = 0;
#pragma unroll
      for (int tj = 0; tj < TILES_; ++tj) {
        const int s = tj * 16 + hl;     // this lane's column in tile tj
        if (s < M_) {
          const float v = acc[tj][r];
          if (v > best) { best = v; bs = s; }   // s ascending -> first occurrence
        }
      }
      // Butterfly reduce across the 16 lanes of each half-wave.
      for (int msk = 8; msk >= 1; msk >>= 1) {
        const float ov = __shfl_xor(best, msk, 32);
        const int   oi = __shfl_xor(bs,   msk, 32);
        if (ov > best || (ov == best && oi < bs)) { best = ov; bs = oi; }
      }
      const int m = ti * 16 + upper * 8 + r;
      if (hl == 0 && m < M_) { mout[m] = best; aout[m] = bs; }
    }
  }
}

// ---------------------------------------------------------------------------
// Kernel 4: per (b,q): mutual-nearest-neighbor masking + predict + NLL.
// Exactly replicates jnp.argmax first-occurrence and where(class_m_max==0).
// ---------------------------------------------------------------------------
__global__ void mn4_finalize(const float* __restrict__ maxv, const int* __restrict__ args,
                             const int* __restrict__ qy, float* __restrict__ lossbuf)
{
  const int bq  = blockIdx.x;
  const int tid = threadIdx.x;
  __shared__ float rowmax[NW_ * M_];
  __shared__ int   argl[NW_ * M_];
  __shared__ int   qidx[M_];
  __shared__ float cwm[M_];
  __shared__ int   snl[NW_ * M_];
  __shared__ float qmask[M_];
  __shared__ float red[256];
  __shared__ float pred[NW_];

  for (int i = tid; i < NW_ * M_; i += 256) {
    rowmax[i] = maxv[(size_t)bq * NW_ * M_ + i];
    argl[i]   = args[(size_t)bq * NW_ * M_ + i];
  }
  __syncthreads();

  // query_nearest (flattened nm = n*M+s, n-major first occurrence) + class_wise_max+1
  if (tid < M_) {
    float best = -INFINITY; int bn = 0, bs = 0;
    for (int nn = 0; nn < NW_; ++nn) {
      const float v = rowmax[nn * M_ + tid];
      if (v > best) { best = v; bn = nn; bs = argl[nn * M_ + tid]; }
    }
    qidx[tid] = bn * M_ + bs;
    cwm[tid]  = best + 1.0f;
  }
  __syncthreads();

  // support_nearest: argmax over m of one_hot(qidx[m]==nm)*cwm[m]; 0-column -> M+1
  for (int nm = tid; nm < NW_ * M_; nm += 256) {
    float best = -INFINITY; int bm = 0;
    for (int m = 0; m < M_; ++m) {
      const float v = (qidx[m] == nm) ? cwm[m] : 0.0f;
      if (v > best) { best = v; bm = m; }
    }
    snl[nm] = (best == 0.0f) ? (M_ + 1) : bm;
  }
  __syncthreads();

  // mutual-NN mask * TEMPERATURE
  if (tid < M_) {
    const int g = snl[qidx[tid]];
    qmask[tid] = (g == tid) ? 2.0f : 0.0f;
  }
  __syncthreads();

  // predict[n] = sum_m rowmax[n][m]*qmask[m]  (deterministic tree reduction)
  for (int nn = 0; nn < NW_; ++nn) {
    red[tid] = (tid < M_) ? rowmax[nn * M_ + tid] * qmask[tid] : 0.f;
    __syncthreads();
    for (int s = 128; s > 0; s >>= 1) {
      if (tid < s) red[tid] += red[tid + s];
      __syncthreads();
    }
    if (tid == 0) pred[nn] = red[0];
    __syncthreads();
  }

  if (tid == 0) {
    float mx = pred[0];
    for (int nn = 1; nn < NW_; ++nn) mx = fmaxf(mx, pred[nn]);
    float se = 0.f;
    for (int nn = 0; nn < NW_; ++nn) se += expf(pred[nn] - mx);
    const float lse = mx + logf(se);
    const int y = qy[bq];
    lossbuf[bq] = lse - pred[y];              // -log_softmax[y]
  }
}

// ---------------------------------------------------------------------------
// Kernel 5: deterministic mean over b*q losses -> scalar
// ---------------------------------------------------------------------------
__global__ void mn4_sum(const float* __restrict__ lossbuf, float* __restrict__ out)
{
  __shared__ float red[256];
  const int tid = threadIdx.x;
  float s = 0.f;
  for (int i = tid; i < B_ * Q_; i += 256) s += lossbuf[i];
  red[tid] = s;
  __syncthreads();
  for (int k = 128; k > 0; k >>= 1) {
    if (tid < k) red[tid] += red[tid + k];
    __syncthreads();
  }
  if (tid == 0) out[0] = red[0] / (float)(B_ * Q_);
}

// ---------------------------------------------------------------------------
// Launch. Workspace layout (~87.6 MB):
//   sup16 : B*NW*MP*C f16   (5.32 MB)
//   qn16  : B*Q *MP*C f16   (79.87 MB)
//   maxv  : B*Q*NW*M  f32   (1.18 MB)
//   args  : B*Q*NW*M  i32   (1.18 MB)
//   loss  : B*Q       f32
// ---------------------------------------------------------------------------
extern "C" void kernel_launch(void* const* d_in, const int* in_sizes, int n_in,
                              void* d_out, int out_size, void* d_ws, size_t ws_size,
                              hipStream_t stream) {
  (void)in_sizes; (void)n_in; (void)out_size; (void)ws_size;
  const float* sup_xf = (const float*)d_in[0];
  // d_in[1] (support_y) unused by the reference loss
  const float* qry_xf = (const float*)d_in[2];
  const int*   qry_y  = (const int*)d_in[3];

  char* ws = (char*)d_ws;
  size_t off = 0;
  _Float16* sup16 = (_Float16*)(ws + off); off += (size_t)B_ * NW_ * MP_ * C_ * sizeof(_Float16);
  _Float16* qn16  = (_Float16*)(ws + off); off += (size_t)B_ * Q_  * MP_ * C_ * sizeof(_Float16);
  float* maxv     = (float*)(ws + off);    off += (size_t)B_ * Q_ * NW_ * M_ * sizeof(float);
  int*   args     = (int*)(ws + off);      off += (size_t)B_ * Q_ * NW_ * M_ * sizeof(int);
  float* lossbuf  = (float*)(ws + off);    off += (size_t)B_ * Q_ * sizeof(float);

  hipLaunchKernelGGL(mn4_prep_support, dim3(B_ * NW_),       dim3(256), 0, stream, sup_xf, sup16);
  hipLaunchKernelGGL(mn4_prep_query,   dim3(B_ * Q_),        dim3(256), 0, stream, qry_xf, qn16);
  hipLaunchKernelGGL(mn4_gemm,         dim3(B_ * Q_ * NW_),  dim3(512), 0, stream, qn16, sup16, maxv, args);
  hipLaunchKernelGGL(mn4_finalize,     dim3(B_ * Q_),        dim3(256), 0, stream, maxv, args, qry_y, lossbuf);
  hipLaunchKernelGGL(mn4_sum,          dim3(1),              dim3(256), 0, stream, lossbuf, (float*)d_out);
}
